// Encoder_36885179138516
// MI455X (gfx1250) — compile-verified, matrix operand order checked by reading, not run on hardware
//
#include <hip/hip_runtime.h>
#include <hip/hip_bf16.h>

// ---------------------------------------------------------------------------
// GCN-VAE encoder for MI455X (gfx1250, wave32).
//   h   = relu( nd * SpMM( (feat*ns) @ W1 ) + b1 )
//   mu  =        nd * SpMM( (h*ns) @ Wmu )  + bmu
//   ls  =        nd * SpMM( (h*ns) @ Wls )  + bls
//   out = mu + noise * exp(ls)
// GEMMs use v_wmma_f32_16x16x32_bf16; the fused mu/ls GEMM stages its A-tiles
// through LDS with global_load_async_to_lds_b128 (ASYNCcnt pipelined).
// ---------------------------------------------------------------------------

typedef __attribute__((ext_vector_type(16))) __bf16 v16bf;
typedef __attribute__((ext_vector_type(8)))  float  v8f;

#define OUTF 128   // output feature width of every layer

// ---------------- degree accumulation (src -> deg_out, dst -> deg_in) -------
__global__ void k_degree(const int* __restrict__ src, const int* __restrict__ dst,
                         float* __restrict__ degs, float* __restrict__ degd, int E) {
    int e = blockIdx.x * blockDim.x + threadIdx.x;
    if (e < E) {
        __hip_atomic_fetch_add(&degs[src[e]], 1.0f, __ATOMIC_RELAXED, __HIP_MEMORY_SCOPE_AGENT);
        __hip_atomic_fetch_add(&degd[dst[e]], 1.0f, __ATOMIC_RELAXED, __HIP_MEMORY_SCOPE_AGENT);
    }
}

// ns = deg_out^-1/2, nd = deg_in^-1/2 (degrees clamped to >= 1)
__global__ void k_norm(float* __restrict__ a, float* __restrict__ b, int N) {
    int i = blockIdx.x * blockDim.x + threadIdx.x;
    if (i < N) {
        a[i] = rsqrtf(fmaxf(a[i], 1.0f));
        b[i] = rsqrtf(fmaxf(b[i], 1.0f));
    }
}

// ---------------- pre-pack W (K x 128, row-major, f32) into bf16 B-fragments
// Fragment f = (colTile*ksteps + ks); per lane 16 bf16 (8 dwords) in the ISA
// 16-bit B layout: lanes 0-15 -> K {0..7,16..23}, lanes 16-31 -> K {8..15,24..31}.
__global__ void k_pack_w(const float* __restrict__ W, unsigned* __restrict__ wf, int ksteps) {
    const int f    = blockIdx.x;
    const int lane = threadIdx.x;          // 0..31
    const int ct   = f / ksteps;
    const int ks   = f % ksteps;
    const int col  = ct * 16 + (lane & 15);
    const int kb   = ks * 32 + ((lane >> 4) ? 8 : 0);
    union { v16bf v; unsigned u[8]; } t;
#pragma unroll
    for (int j = 0; j < 8; ++j) {
        t.v[j]     = (__bf16)W[(size_t)(kb + j)      * OUTF + col];
        t.v[8 + j] = (__bf16)W[(size_t)(kb + 16 + j) * OUTF + col];
    }
    unsigned* dstp = wf + ((size_t)f * 32 + lane) * 8;
#pragma unroll
    for (int j = 0; j < 8; ++j) dstp[j] = t.u[j];
}

// ---------------- layer-1 GEMM: C[N,128] = bf16(feat*ns) @ W1 ---------------
// block = 256 threads (8 waves); block -> 16-row strip; wave w -> cols [16w,16w+16)
__global__ void __launch_bounds__(256)
k_gemm_bf16(const float* __restrict__ A, const float* __restrict__ ns,
            const unsigned* __restrict__ wf, float* __restrict__ C,
            int N, int K) {
    __shared__ __bf16 As[16][32];          // one 16x32 bf16 A-tile (1 KB)

    const int tid  = threadIdx.x;
    const int wave = tid >> 5;
    const int lane = tid & 31;
    const int l15  = lane & 15;
    const int half = lane >> 4;            // 0 or 1
    const int rowBase = blockIdx.x * 16;
    const int ksteps  = K >> 5;

    v8f acc = {0.f, 0.f, 0.f, 0.f, 0.f, 0.f, 0.f, 0.f};

    for (int ks = 0; ks < ksteps; ++ks) {
        // cooperative load + scale + f32->bf16 of the A tile (512 elems / 256 thr)
#pragma unroll
        for (int i = 0; i < 2; ++i) {
            int idx = tid + i * 256;       // 0..511
            int r = idx >> 5, c = idx & 31;
            int row = rowBase + r;
            if (row >= N) row = N - 1;
            As[r][c] = (__bf16)(A[(size_t)row * K + ks * 32 + c] * ns[row]);
        }
        if (ks + 1 < ksteps)               // pull next A tile toward L2/L0
            __builtin_prefetch(A + (size_t)(rowBase + l15) * K + (ks + 1) * 32, 0, 1);
        __syncthreads();

        // A fragment from LDS (two contiguous 16B reads per lane)
        union { v16bf v; unsigned u[8]; } af;
        const unsigned* Ar = (const unsigned*)&As[l15][0];
        const int ub = half * 4;
#pragma unroll
        for (int i = 0; i < 4; ++i) {
            af.u[i]     = Ar[ub + i];       // K = base .. base+7
            af.u[4 + i] = Ar[ub + 8 + i];   // K = base+16 .. base+23
        }

        // B fragment: prepacked, one contiguous 32B load per lane
        union { v16bf v; unsigned u[8]; } bfr;
        const unsigned* bp = wf + (((size_t)wave * ksteps + ks) * 32 + lane) * 8;
#pragma unroll
        for (int i = 0; i < 8; ++i) bfr.u[i] = bp[i];

        acc = __builtin_amdgcn_wmma_f32_16x16x32_bf16(
                  false, af.v, false, bfr.v, (short)0, acc, false, false);
        __syncthreads();
    }

    const int col = wave * 16 + l15;
    const int rb  = rowBase + half * 8;
#pragma unroll
    for (int r = 0; r < 8; ++r) {
        int row = rb + r;
        if (row < N) C[(size_t)row * OUTF + col] = acc[r];
    }
}

// ---------------- Hb = bf16(H * ns[row])  (pre-scaled A for mu/ls GEMMs) ----
__global__ void k_scale_bf16(const float* __restrict__ H, const float* __restrict__ ns,
                             __bf16* __restrict__ Hb, int N) {
    long long i = (long long)blockIdx.x * blockDim.x + threadIdx.x;
    if (i >= (long long)N * OUTF) return;
    int r = (int)(i >> 7);
    Hb[i] = (__bf16)(H[i] * ns[r]);
}

// ---------------- fused mu/ls GEMM with async-LDS A-tile pipeline -----------
// block = 512 threads (16 waves) per 16-row strip of Ab (bf16, K=128).
// waves 0-7 -> Wmu col tiles into Ca; waves 8-15 -> Wls col tiles into Cb.
// Wave 0 DMA-copies the 1 KB A-tile into double-buffered LDS with
// global_load_async_to_lds_b128, one k-step ahead (ASYNCcnt + barrier sync).
__global__ void __launch_bounds__(512)
k_gemm_dual_bf16(const __bf16* __restrict__ Ab,
                 const unsigned* __restrict__ wfa, const unsigned* __restrict__ wfb,
                 float* __restrict__ Ca, float* __restrict__ Cb, int N) {
    const int K = OUTF;                    // 128
    const int ksteps = K >> 5;             // 4
    __shared__ __bf16 As[2][16][32];       // 2 x 1 KB double buffer

    const int tid  = threadIdx.x;
    const int wave = tid >> 5;
    const int lane = tid & 31;
    const int l15  = lane & 15;
    const int half = lane >> 4;
    const int rowBase = blockIdx.x * 16;

    const unsigned* wf = (wave < 8) ? wfa : wfb;
    float*          C  = (wave < 8) ? Ca  : Cb;
    const int       ct = wave & 7;

    // issue async copy of a 16x32 bf16 tile (64 chunks of 16B, 32 lanes x 2)
    auto issue_tile = [&](int buf, int ks) {
#pragma unroll
        for (int j = 0; j < 2; ++j) {
            int chunk = lane + 32 * j;          // 0..63
            int r  = chunk >> 2;                // tile row 0..15
            int co = (chunk & 3) * 8;           // col offset 0,8,16,24
            int row = rowBase + r;
            if (row >= N) row = N - 1;
            unsigned    ldso = (unsigned)(unsigned long long)&As[buf][r][co];
            const void* gp   = Ab + (size_t)row * K + ks * 32 + co;
            asm volatile("global_load_async_to_lds_b128 %0, %1, off"
                         :: "v"(ldso), "v"(gp) : "memory");
        }
    };

    if (wave == 0) issue_tile(0, 0);

    v8f acc = {0.f, 0.f, 0.f, 0.f, 0.f, 0.f, 0.f, 0.f};

    for (int ks = 0; ks < ksteps; ++ks) {
        const int cur = ks & 1;
        if (wave == 0)
            asm volatile("s_wait_asynccnt 0x0" ::: "memory");
        __syncthreads();                        // tile `cur` visible to all waves
        if (wave == 0 && ks + 1 < ksteps)
            issue_tile(cur ^ 1, ks + 1);        // DMA next tile while computing

        union { v16bf v; unsigned u[8]; } af;
        const unsigned* Ar = (const unsigned*)&As[cur][l15][0];
        const int ub = half * 4;
#pragma unroll
        for (int i = 0; i < 4; ++i) {
            af.u[i]     = Ar[ub + i];
            af.u[4 + i] = Ar[ub + 8 + i];
        }

        union { v16bf v; unsigned u[8]; } bfr;
        const unsigned* bp = wf + (((size_t)ct * ksteps + ks) * 32 + lane) * 8;
#pragma unroll
        for (int i = 0; i < 8; ++i) bfr.u[i] = bp[i];

        acc = __builtin_amdgcn_wmma_f32_16x16x32_bf16(
                  false, af.v, false, bfr.v, (short)0, acc, false, false);
        __syncthreads();                        // readers done before buffer reuse
    }

    const int col = ct * 16 + l15;
    const int rb  = rowBase + half * 8;
#pragma unroll
    for (int r = 0; r < 8; ++r) {
        int row = rb + r;
        if (row < N) C[(size_t)row * OUTF + col] = acc[r];
    }
}

// ---------------- SpMM scatter: Y[dst] += X[src]  (float4 per thread) -------
__global__ void k_spmm(const float* __restrict__ X, const int* __restrict__ src,
                       const int* __restrict__ dst, float* __restrict__ Y, int E) {
    long long gid = (long long)blockIdx.x * blockDim.x + threadIdx.x;
    if (gid >= (long long)E * 32) return;
    int e = (int)(gid >> 5);
    int q = ((int)gid & 31) * 4;
    int s = src[e], d = dst[e];
    const float4 v = *(const float4*)(X + (size_t)s * OUTF + q);
    float* yp = Y + (size_t)d * OUTF + q;
    __hip_atomic_fetch_add(yp + 0, v.x, __ATOMIC_RELAXED, __HIP_MEMORY_SCOPE_AGENT);
    __hip_atomic_fetch_add(yp + 1, v.y, __ATOMIC_RELAXED, __HIP_MEMORY_SCOPE_AGENT);
    __hip_atomic_fetch_add(yp + 2, v.z, __ATOMIC_RELAXED, __HIP_MEMORY_SCOPE_AGENT);
    __hip_atomic_fetch_add(yp + 3, v.w, __ATOMIC_RELAXED, __HIP_MEMORY_SCOPE_AGENT);
}

// ---------------- h = relu(h * nd[row] + b[col]) ----------------------------
__global__ void k_bias_relu(float* __restrict__ H, const float* __restrict__ nd,
                            const float* __restrict__ b, int N) {
    long long i = (long long)blockIdx.x * blockDim.x + threadIdx.x;
    if (i >= (long long)N * OUTF) return;
    int r = (int)(i >> 7), c = (int)i & 127;
    float v = H[i] * nd[r] + b[c];
    H[i] = v > 0.f ? v : 0.f;
}

// ---------------- out = (amu*nd+bmu) + noise * exp(als*nd+bls) --------------
__global__ void k_final(float* __restrict__ out, const float* __restrict__ als,
                        const float* __restrict__ nd, const float* __restrict__ bmu,
                        const float* __restrict__ bls, const float* __restrict__ noise,
                        int N) {
    long long i = (long long)blockIdx.x * blockDim.x + threadIdx.x;
    if (i >= (long long)N * OUTF) return;
    int r = (int)(i >> 7), c = (int)i & 127;
    float mu = out[i] * nd[r] + bmu[c];
    float ls = als[i] * nd[r] + bls[c];
    out[i] = mu + noise[i] * expf(ls);
}

// ---------------------------------------------------------------------------
extern "C" void kernel_launch(void* const* d_in, const int* in_sizes, int n_in,
                              void* d_out, int out_size, void* d_ws, size_t ws_size,
                              hipStream_t stream) {
    const float* feat  = (const float*)d_in[0];
    const int*   edges = (const int*)  d_in[1];
    const float* W1    = (const float*)d_in[2];
    const float* b1    = (const float*)d_in[3];
    const float* Wmu   = (const float*)d_in[4];
    const float* bmu   = (const float*)d_in[5];
    const float* Wls   = (const float*)d_in[6];
    const float* bls   = (const float*)d_in[7];
    const float* noise = (const float*)d_in[8];

    const int IN = 512;
    const int N  = in_sizes[0] / IN;
    const int E  = in_sizes[1] / 2;
    const int* src = edges;
    const int* dst = edges + E;

    // ---- workspace carve-up (~181 MB) ----
    char* ws = (char*)d_ws;
    auto al = [](size_t x) { return (x + 255) & ~(size_t)255; };
    size_t off = 0;
    float* ns = (float*)(ws + off); off = al(off + (size_t)N * 4);
    float* nd = (float*)(ws + off); off = al(off + (size_t)N * 4);
    float* t1 = (float*)(ws + off); off = al(off + (size_t)N * OUTF * 4);
    float* h  = (float*)(ws + off); off = al(off + (size_t)N * OUTF * 4);
    float* t2 = (float*)(ws + off); off = al(off + (size_t)N * OUTF * 4);
    __bf16* hb = (__bf16*)(ws + off); off = al(off + (size_t)N * OUTF * 2);
    unsigned* wf1 = (unsigned*)(ws + off); off = al(off + (size_t)8 * (IN   / 32) * 32 * 32);
    unsigned* wf2 = (unsigned*)(ws + off); off = al(off + (size_t)8 * (OUTF / 32) * 32 * 32);
    unsigned* wf3 = (unsigned*)(ws + off); off = al(off + (size_t)8 * (OUTF / 32) * 32 * 32);

    float* amu = (float*)d_out;                 // mu aggregation target == output

    const int TPB = 256;
    const int gEdge = (E + TPB - 1) / TPB;
    const int gNode = (N + TPB - 1) / TPB;
    const int gRowT = (N + 15) / 16;
    const long long elems = (long long)N * OUTF;
    const int gElem = (int)((elems + TPB - 1) / TPB);
    const long long sp = (long long)E * 32;
    const int gSpmm = (int)((sp + TPB - 1) / TPB);

    // 1) symmetric degree normalizers
    hipMemsetAsync(ns, 0, (size_t)N * 4, stream);
    hipMemsetAsync(nd, 0, (size_t)N * 4, stream);
    k_degree<<<gEdge, TPB, 0, stream>>>(src, dst, ns, nd, E);
    k_norm<<<gNode, TPB, 0, stream>>>(ns, nd, N);

    // 2) pre-pack weights into WMMA B-fragment order (bf16)
    k_pack_w<<<8 * (IN   / 32), 32, 0, stream>>>(W1,  wf1, IN   / 32);
    k_pack_w<<<8 * (OUTF / 32), 32, 0, stream>>>(Wmu, wf2, OUTF / 32);
    k_pack_w<<<8 * (OUTF / 32), 32, 0, stream>>>(Wls, wf3, OUTF / 32);

    // 3) layer 1: project (WMMA), aggregate (atomics), bias+relu
    k_gemm_bf16<<<gRowT, TPB, 0, stream>>>(feat, ns, wf1, t1, N, IN);
    hipMemsetAsync(h, 0, (size_t)N * OUTF * 4, stream);
    k_spmm<<<gSpmm, TPB, 0, stream>>>(t1, src, dst, h, E);
    k_bias_relu<<<gElem, TPB, 0, stream>>>(h, nd, b1, N);

    // 4) fused mu/ls projections: pre-scale h to bf16, then one async-pipelined
    //    dual-weight WMMA GEMM (reads h once instead of twice)
    k_scale_bf16<<<gElem, TPB, 0, stream>>>(h, ns, hb, N);
    k_gemm_dual_bf16<<<gRowT, 512, 0, stream>>>(hb, wf2, wf3, t1, t2, N);

    // 5) aggregate mu -> d_out, ls -> h (h free after hb was built)
    hipMemsetAsync(amu, 0, (size_t)N * OUTF * 4, stream);
    k_spmm<<<gSpmm, TPB, 0, stream>>>(t1, src, dst, amu, E);
    hipMemsetAsync(h, 0, (size_t)N * OUTF * 4, stream);
    k_spmm<<<gSpmm, TPB, 0, stream>>>(t2, src, dst, h, E);

    // 6) reparameterization epilogue
    k_final<<<gElem, TPB, 0, stream>>>(amu, h, nd, bmu, bls, noise, N);
}
